// PaiConvMax_91285234909706
// MI455X (gfx1250) — matrix operand order repaired and checked by reading, not compiled, not run on hardware
//
#include <hip/hip_runtime.h>
#include <stdint.h>

// ---------------- types ----------------
typedef _Float16 v16h __attribute__((ext_vector_type(16)));
typedef float    v8f  __attribute__((ext_vector_type(8)));
typedef unsigned int u32x4 __attribute__((ext_vector_type(4)));
typedef int      i32x4 __attribute__((ext_vector_type(4)));
typedef int      i32x8 __attribute__((ext_vector_type(8)));

union HFrag { v16h v; _Float16 h[16]; };

// ---------------- problem constants ----------------
#define Cc   64
#define Nn   2048
#define Kk   20
#define NWG  256
#define TILES_PER_WG 8   // 256 WG * 8 tiles * 16 pts = 32768

// ---------------- ws layout (bytes) ----------------
#define WS_W1H   0            // 81920 f16 = 163840 B (W1 B-fragments, pre-swizzled)
#define WS_W2H   163840       // 8192  f16 = 16384 B  (full W2 A-fragments, K=128)
#define WS_PART  180224       // 256 WG * 256 f32 = 262144 B (deterministic partials)
#define WS_SS    442368       // 256 f32 (scale, shift)

#if defined(__HIP_DEVICE_COMPILE__) && __has_builtin(__builtin_amdgcn_tensor_load_to_lds)
#define HAS_TDM 1
#else
#define HAS_TDM 0
#endif

// A-fragment (16-bit, 16x32) K index for (lane, half-element j)
__device__ __forceinline__ int kmapA(int lane, int j) {
  return ((j < 8) ? j : j + 8) + ((lane & 16) ? 8 : 0);
}

// inverse map: flat K index kk + fragment row -> half-element slot inside
// an A-fragment array laid out as [(ks*32 + lane) * 16 + j]
__device__ __forceinline__ int fragSlot(int kk, int row) {
  const int ks = kk >> 5, r5 = kk & 31;
  const int lh = (r5 >> 3) & 1;
  const int j  = (r5 & 7) + ((r5 & 16) ? 8 : 0);
  return ((ks * 32 + lh * 16 + row) << 4) + j;
}

// ============ prep: pre-swizzle W1 / W2 fragments into ws (f16) ============
__global__ __launch_bounds__(256)
void paiconv_prep(const float* __restrict__ W1, const float* __restrict__ W2,
                  _Float16* __restrict__ wsW1h, _Float16* __restrict__ wsW2h)
{
  const int idx = blockIdx.x * 256 + threadIdx.x;
  if (idx < 81920) {
    // W1 as B-operand: frag f = ks*4+nt; b[j] = W1T[ks*32 + kmapA(lane,j)][nt*16 + l15]
    const int f = idx >> 9, rem = idx & 511, lane = rem >> 4, j = rem & 15;
    const int ks = f >> 2, nt = f & 3;
    const int o  = nt * 16 + (lane & 15);
    const int kk = ks * 32 + kmapA(lane, j);
    wsW1h[idx] = (_Float16)W1[o * 1280 + kk];
  } else if (idx < 90112) {
    // full W2 (64 x 128) as A-operand: frag f = ks*4+mt (ks 0..3)
    const int q = idx - 81920;
    const int f = q >> 9, rem = q & 511, lane = rem >> 4, j = rem & 15;
    const int ks = f >> 2, mt = f & 3;
    const int o  = mt * 16 + (lane & 15);
    const int cc = ks * 32 + ((lane & 16) ? 16 : 0) + j;
    wsW2h[q] = (_Float16)W2[o * 128 + cc];
  }
}

// ============ main kernel: gather + 3 WMMA GEMMs + epilogue ============
__global__ __launch_bounds__(256)
void paiconv_main(const float* __restrict__ feature,
                  const int*   __restrict__ neigh,
                  const float* __restrict__ permat,
                  const float* __restrict__ b1g,
                  const float* __restrict__ b2g,
                  const _Float16* __restrict__ wsW1h,
                  const _Float16* __restrict__ wsW2h,
                  float* __restrict__ wsPart,
                  float* __restrict__ out)
{
  __shared__ _Float16 sFG[32768] __attribute__((aligned(32)));  // F A-frags -> per-point (G-f0) W2-B frags
  __shared__ _Float16 sGfl[20480] __attribute__((aligned(32))); // Gflat in W1 A-frag layout (rows=points)
  __shared__ float    sPraw[6400];  // permatrix tile (16 x 20 x 20) f32, TDM destination
  __shared__ int      sQ[320];      // neighbor indices for tile
  __shared__ _Float16 sG0h[1024];   // G[:,0] per point (f16)
  __shared__ float    sO1[1024];    // out1 staging (no bias)
  __shared__ float    sO2[1024];    // out2 staging (no bias)

  const int tid  = threadIdx.x;
  const int lane = tid & 31;
  const int wid  = tid >> 5;
  const int l15  = lane & 15;
  const int prow = (lane & 16) ? 8 : 0;
  const int khalfB = (lane & 16) ? 16 : 0;

  float bias = 0.f;
  if (tid < 128) bias = (tid < 64) ? b1g[tid] : b2g[tid - 64];

  float accS = 0.f, accQ = 0.f;

  for (int it = 0; it < TILES_PER_WG; ++it) {
    const int tileBase = (blockIdx.x * TILES_PER_WG + it) * 16;
    __syncthreads();   // protect LDS reuse across iterations

    // ---- phase B: zero F frags, load neighbor ids, kick TDM for permatrix tile ----
    #pragma unroll 4
    for (int i = tid; i < 16384; i += 256) ((unsigned*)sFG)[i] = 0u;
    for (int i = tid; i < 320; i += 256) sQ[i] = neigh[tileBase * Kk + i];

#if HAS_TDM
    if (wid == 0) {
      const uint64_t ga = (uint64_t)(uintptr_t)(permat + (size_t)tileBase * 400);
      const unsigned lds = (unsigned)(uintptr_t)(&sPraw[0]);
      u32x4 g0 = { 1u,                                    // count=1, user mode
                   lds,                                   // lds_addr
                   (unsigned)(ga & 0xFFFFFFFFu),          // global_addr lo
                   (unsigned)((ga >> 32) & 0x01FFFFFFu) | (2u << 30) }; // addr hi | type=2
      i32x8 g1 = { 0x20000,                    // data_size = 4B
                   (int)(6400u << 16),         // tensor_dim0 lo16
                   (int)(1u << 16),            // dim0 hi16=0 | tensor_dim1=1
                   (int)(6400u << 16),         // dim1 hi16=0 | tile_dim0=6400
                   1,                          // tile_dim1=1, tile_dim2=0
                   6400,                       // tensor_dim0_stride lo32
                   0, 0 };
      i32x4 z4 = { 0, 0, 0, 0 };
#if __clang_major__ >= 23
      i32x8 z8 = { 0, 0, 0, 0, 0, 0, 0, 0 };
      __builtin_amdgcn_tensor_load_to_lds(g0, g1, z4, z4, z8, 0);
#else
      __builtin_amdgcn_tensor_load_to_lds(g0, g1, z4, z4, 0);
#endif
    }
#endif
    __syncthreads();

    // ---- phase C: gather F into A-fragment layout (overlaps TDM) ----
    for (int e = tid; e < 20480; e += 256) {
      const int cp = e & 63;          // shuffled channel c'
      const int t2 = e >> 6;          // p*20 + k
      const int p  = t2 / Kk;
      const int k  = t2 - p * Kk;
      const int q  = sQ[p * Kk + k];
      const int qb = q >> 11, qn = q & 2047;
      const int c  = ((cp & 3) << 4) | (cp >> 2);   // inverse group shuffle
      const float v = feature[(qb * Cc + c) * Nn + qn];
      const int f = p * 4 + (cp >> 4);
      const int lanehi = (k >> 3) & 1;
      const int j = (k & 7) + ((k >= 16) ? 8 : 0);
      sFG[(f * 32 + lanehi * 16 + (cp & 15)) * 16 + j] = (_Float16)v;
    }
#if HAS_TDM
    if (wid == 0) __builtin_amdgcn_s_wait_tensorcnt(0);
#else
    for (int i = tid; i < 6400; i += 256) sPraw[i] = permat[(size_t)tileBase * 400 + i];
#endif
    __syncthreads();

    // ---- phase G: per-point G = F (64x20) @ P (20x20), 2 points per wave ----
    #pragma unroll
    for (int pp = 0; pp < 2; ++pp) {
      const int p = wid * 2 + pp;
      HFrag A[4];
      #pragma unroll
      for (int mt = 0; mt < 4; ++mt)
        A[mt].v = *(const v16h*)&sFG[((p * 4 + mt) * 32 + lane) * 16];
      HFrag Bp[2];
      #pragma unroll
      for (int nt = 0; nt < 2; ++nt) {
        const int m  = nt * 16 + l15;
        const int mC = (m < Kk) ? m : (Kk - 1);      // clamp: unconditional LDS load
        #pragma unroll
        for (int j = 0; j < 16; ++j) {
          const int kk  = ((j < 8) ? j : j + 8) + prow;
          const int kkC = (kk < Kk) ? kk : (Kk - 1);
          const float v = sPraw[p * 400 + kkC * Kk + mC];
          Bp[nt].h[j] = (_Float16)((kk < Kk && m < Kk) ? v : 0.0f);  // v_cndmask
        }
      }
      v8f Cg[8];
      #pragma unroll
      for (int mt = 0; mt < 4; ++mt)
        #pragma unroll
        for (int nt = 0; nt < 2; ++nt) {
          v8f c = {0.f,0.f,0.f,0.f,0.f,0.f,0.f,0.f};
          Cg[mt * 2 + nt] = __builtin_amdgcn_wmma_f32_16x16x32_f16(
              false, A[mt].v, false, Bp[nt].v, (short)0, c, false, false);
        }
      // store G in consumer fragment layouts:
      //  (a) sGfl: Gflat (K = c*20+k) in W1 A-frag order, rows = points
      //  (b) sFG[p region]: (G - f0) in W2 B-frag order (cols >=20 are masked later)
      #pragma unroll
      for (int mt = 0; mt < 4; ++mt)
        #pragma unroll
        for (int i = 0; i < 8; ++i) {
          const int crow = mt * 16 + i + prow;
          const float v0 = Cg[mt * 2 + 0][i];           // col = l15
          const float v1 = Cg[mt * 2 + 1][i];           // col = 16 + l15
          const float g0 = __shfl(v0, 0, 16);           // f0[crow] broadcast in 16-group
          sGfl[fragSlot(crow * Kk + l15, p)] = (_Float16)v0;
          const int ksb   = crow >> 5;
          const int inner = crow & 31;
          const int base  = p * 2048 +
                            ((ksb * 2) * 32 + (inner >> 4) * 16 + l15) * 16 + (inner & 15);
          sFG[base]       = (_Float16)(v0 - g0);        // nt2 = 0 frag
          sFG[base + 512] = (_Float16)(v1 - g0);        // nt2 = 1 frag (+32*16 halfs)
        }
      if (l15 < 4) {                                    // cols 16..19 of Gflat (single branch)
        #pragma unroll
        for (int mt = 0; mt < 4; ++mt)
          #pragma unroll
          for (int i = 0; i < 8; ++i) {
            const int crow = mt * 16 + i + prow;
            sGfl[fragSlot(crow * Kk + 16 + l15, p)] = (_Float16)Cg[mt * 2 + 1][i];
          }
      }
    }
    __syncthreads();

    // ---- extract G[:,0] (f0) unconditionally from sGfl ----
    #pragma unroll
    for (int i = tid; i < 1024; i += 256) {
      const int p = i >> 6, c = i & 63;
      sG0h[i] = sGfl[fragSlot(c * Kk, p)];
    }
    __syncthreads();

    // ---- phase E: waves 0-3 -> W1 GEMM; waves 4-7 -> W2 @ cat GEMM + max ----
    if (wid < 4) {
      const int nt = wid;
      v8f c = {0.f,0.f,0.f,0.f,0.f,0.f,0.f,0.f};
      #pragma unroll 4
      for (int ks = 0; ks < 40; ++ks) {
        const v16h Av = *(const v16h*)&sGfl[(ks * 32 + lane) * 16];               // contiguous
        const v16h Bv = *(const v16h*)(wsW1h + ((ks * 4 + nt) * 32 + lane) * 16); // contiguous
        c = __builtin_amdgcn_wmma_f32_16x16x32_f16(false, Av, false, Bv,
                                                   (short)0, c, false, false);
      }
      const int o = nt * 16 + l15;
      #pragma unroll
      for (int i = 0; i < 8; ++i) sO1[(i + prow) * 64 + o] = c[i];
    } else {
      for (int pi = 0; pi < 4; ++pi) {
        const int p = (wid - 4) * 4 + pi;
        // B = cat (K=128): rows 0..63 = f0 broadcast, rows 64..127 = G - f0
        HFrag Bf[6];   // [0..1]: f0 halves (col-independent); [2..5]: (nt,ks) contiguous
        #pragma unroll
        for (int ks = 0; ks < 2; ++ks) {
          const int cb = ks * 32 + khalfB;
          #pragma unroll
          for (int j = 0; j < 16; ++j)
            Bf[ks].h[j] = sG0h[p * 64 + cb + j];          // LDS broadcast
        }
        #pragma unroll
        for (int nt = 0; nt < 2; ++nt)
          #pragma unroll
          for (int ks = 0; ks < 2; ++ks)
            Bf[2 + nt * 2 + ks].v =
                *(const v16h*)&sFG[p * 2048 + ((ks * 2 + nt) * 32 + lane) * 16];
        #pragma unroll
        for (int mt = 0; mt < 4; ++mt) {
          v8f c0 = {0.f,0.f,0.f,0.f,0.f,0.f,0.f,0.f};
          v8f c1 = {0.f,0.f,0.f,0.f,0.f,0.f,0.f,0.f};
          #pragma unroll
          for (int ks = 0; ks < 4; ++ks) {
            const v16h Av = *(const v16h*)(wsW2h + ((ks * 4 + mt) * 32 + lane) * 16);
            const v16h B0 = (ks < 2) ? Bf[ks].v : Bf[2 + 0 * 2 + (ks - 2)].v;
            const v16h B1 = (ks < 2) ? Bf[ks].v : Bf[2 + 1 * 2 + (ks - 2)].v;
            c0 = __builtin_amdgcn_wmma_f32_16x16x32_f16(false, Av, false, B0,
                                                        (short)0, c0, false, false);
            c1 = __builtin_amdgcn_wmma_f32_16x16x32_f16(false, Av, false, B1,
                                                        (short)0, c1, false, false);
          }
          #pragma unroll
          for (int i = 0; i < 8; ++i) {
            float m = c0[i];
            const float v1 = (l15 < 4) ? c1[i] : -3.0e38f;   // mask cols 20..31
            m = fmaxf(m, v1);
            m = fmaxf(m, __shfl_xor(m, 1, 16));
            m = fmaxf(m, __shfl_xor(m, 2, 16));
            m = fmaxf(m, __shfl_xor(m, 4, 16));
            m = fmaxf(m, __shfl_xor(m, 8, 16));
            if (l15 == 0) sO2[p * 64 + mt * 16 + i + prow] = m;
          }
        }
      }
    }
    __syncthreads();

    // ---- epilogue: store un-normalized output + accumulate channel stats ----
    if (tid < 128) {
      const int ch = tid;
      const float* src = (ch < 64) ? (sO1 + ch) : (sO2 + (ch - 64));
      const int b  = tileBase >> 11;
      const int n0 = tileBase & 2047;
      float* dst = out + ((size_t)(b * 128 + ch)) * Nn + n0;
      #pragma unroll
      for (int p = 0; p < 16; ++p) {
        const float v = src[p * 64] + bias;
        accS += v; accQ += v * v;
        dst[p] = v;
      }
    }
  }

  // deterministic partial sums: one slot per (workgroup, channel)
  if (tid < 128) {
    wsPart[blockIdx.x * 256 + tid]       = accS;
    wsPart[blockIdx.x * 256 + 128 + tid] = accQ;
  }
}

// ============ BN stats reduce (deterministic) ============
__global__ void paiconv_stats(const float* __restrict__ wsPart,
                              const float* __restrict__ gamma,
                              const float* __restrict__ beta,
                              float* __restrict__ wsSS)
{
  const int ch = threadIdx.x;
  if (ch < 128) {
    float s = 0.f, q = 0.f;
    for (int wg = 0; wg < NWG; ++wg) {
      s += wsPart[wg * 256 + ch];
      q += wsPart[wg * 256 + 128 + ch];
    }
    const float mean = s * (1.0f / 32768.0f);
    const float var  = q * (1.0f / 32768.0f) - mean * mean;
    const float sc   = gamma[ch] * rsqrtf(var + 1e-5f);
    wsSS[ch]       = sc;
    wsSS[128 + ch] = beta[ch] - mean * sc;
  }
}

// ============ BN apply, in place, float4 ============
__global__ __launch_bounds__(256)
void paiconv_apply(float* __restrict__ out, const float* __restrict__ wsSS)
{
  const int i4 = blockIdx.x * 256 + threadIdx.x;       // 4 floats each, 2048%4==0
  const int ch = (i4 >> 9) & 127;
  const float sc = wsSS[ch], sh = wsSS[128 + ch];
  float4* p = (float4*)out + i4;
  float4 v = *p;
  v.x = fmaf(v.x, sc, sh);
  v.y = fmaf(v.y, sc, sh);
  v.z = fmaf(v.z, sc, sh);
  v.w = fmaf(v.w, sc, sh);
  *p = v;
}

// ============ launcher ============
extern "C" void kernel_launch(void* const* d_in, const int* in_sizes, int n_in,
                              void* d_out, int out_size, void* d_ws, size_t ws_size,
                              hipStream_t stream) {
  (void)in_sizes; (void)n_in; (void)out_size; (void)ws_size;
  const float* feature = (const float*)d_in[0];
  const int*   neigh   = (const int*)d_in[1];
  const float* permat  = (const float*)d_in[2];
  const float* W1      = (const float*)d_in[3];
  const float* b1      = (const float*)d_in[4];
  const float* W2      = (const float*)d_in[5];
  const float* b2      = (const float*)d_in[6];
  const float* gamma   = (const float*)d_in[7];
  const float* beta    = (const float*)d_in[8];
  float* out = (float*)d_out;

  char* ws = (char*)d_ws;
  _Float16* wsW1h = (_Float16*)(ws + WS_W1H);
  _Float16* wsW2h = (_Float16*)(ws + WS_W2H);
  float* wsPart = (float*)(ws + WS_PART);
  float* wsSS   = (float*)(ws + WS_SS);

  paiconv_prep<<<352, 256, 0, stream>>>(W1, W2, wsW1h, wsW2h);
  paiconv_main<<<NWG, 256, 0, stream>>>(feature, neigh, permat, b1, b2,
                                        wsW1h, wsW2h, wsPart, out);
  paiconv_stats<<<1, 128, 0, stream>>>(wsPart, gamma, beta, wsSS);
  paiconv_apply<<<4096, 256, 0, stream>>>(out, wsSS);
}
